// ResidualAttentionBlock_63496796504673
// MI455X (gfx1250) — matrix operand order repaired and checked by reading, not compiled, not run on hardware
//
#include <hip/hip_runtime.h>

typedef __attribute__((ext_vector_type(16))) _Float16 v16h;
typedef __attribute__((ext_vector_type(8)))  float    v8f;

#define BS    32
#define NTOK  512
#define DM    96
#define NH    4
#define A2LEN 150
#define RMAX  32
#define MTOT  (BS * NTOK)   // 16384 tokens

__device__ __forceinline__ float silu_gate(float v) {
  // v * sigmoid(1.702 * v)
  return v / (1.0f + __expf(-1.702f * v));
}

// ---------------------------------------------------------------------------
// WMMA GEMM: C[m,n] = sum_k A[m,k] * W[n,k] + W[n,K]  (+ resid[m,n] if RESID)
// One wave computes one 16x16 output tile; K consumed 32 at a time via
// v_wmma_f32_16x16x32_f16 (f16 inputs, f32 accumulate).
//
// A-frag layout (ISA 16-bit A 16x32): lane = (m&15) + 16*hi,
//   element j   (0..7)  <- K = k0 + hi*8 + j
//   element j+8 (0..7)  <- K = k0 + 16 + hi*8 + j
// B-frag layout (ISA 16-bit B 32x16): lane = (n&15) + 16*hi,
//   element j   (0..7)  <- K = k0 + hi*8 + j
//   element j+8 (0..7)  <- K = k0 + 16 + hi*8 + j
// C/D layout: acc[v] -> m = mt*16 + v + hi*8, n = nt*16 + (lane&15)
// ---------------------------------------------------------------------------
template<int K, int KSTART, int LDA, int LDW, int LDC, bool GATE, bool RESID, bool QKV_SKIP>
__global__ void gemm_wmma_kernel(const float* __restrict__ A,
                                 const float* __restrict__ W,     // (N, LDW), bias at col K
                                 const float* __restrict__ resid, // pitch LDC (or null)
                                 float* __restrict__ C,
                                 int ntiles)                      // logical N tiles per M tile
{
  const int wave = (int)((blockIdx.x * blockDim.x + threadIdx.x) >> 5);
  const int lane = (int)(threadIdx.x & 31);
  const int mt   = wave / ntiles;
  int       nt   = wave - mt * ntiles;
  if (QKV_SKIP) {
    // map 12 useful tiles -> {2,3,4,5, 8,9,10,11, 14,15,16,17} of 18
    nt = (nt >> 2) * 6 + 2 + (nt & 3);
  }
  const int mlo = lane & 15;
  const int hi  = lane >> 4;
  const int row = mt * 16 + mlo;   // M index for A loads
  const int col = nt * 16 + mlo;   // N index for B loads / C stores

  v8f acc = {};
  #pragma unroll
  for (int k0 = KSTART; k0 < K; k0 += 32) {
    const float* ap0 = A + (size_t)row * LDA + k0 + hi * 8;
    const float* ap1 = ap0 + 16;
    v16h a;
    #pragma unroll
    for (int j = 0; j < 8; ++j) {
      float v0 = ap0[j];
      float v1 = ap1[j];
      if (GATE) { v0 = silu_gate(v0); v1 = silu_gate(v1); }
      a[j]     = (_Float16)v0;
      a[j + 8] = (_Float16)v1;
    }
    const float* bp = W + (size_t)col * LDW + k0 + hi * 8;
    v16h b;
    #pragma unroll
    for (int j = 0; j < 8; ++j) { b[j] = (_Float16)bp[j]; b[j + 8] = (_Float16)bp[j + 16]; }
    acc = __builtin_amdgcn_wmma_f32_16x16x32_f16(
        /*neg_a=*/false, a, /*neg_b=*/false, b,
        /*c_mod=*/(short)0, acc, /*reuse_a=*/false, /*reuse_b=*/false);
  }

  const float bias = W[(size_t)col * LDW + K];
  #pragma unroll
  for (int v = 0; v < 8; ++v) {
    const int m = mt * 16 + v + hi * 8;
    float val = acc[v] + bias;
    if (RESID) val += resid[(size_t)m * LDC + col];
    C[(size_t)m * LDC + col] = val;
  }
}

// ---------------------------------------------------------------------------
// Zero the dense-attention output channels [72,96) of yb for every token.
// (a2a kernel then scatters its 150 tokens per batch on top.)
// ---------------------------------------------------------------------------
__global__ void zero_dense_kernel(float* __restrict__ yb)
{
  const int i = blockIdx.x * blockDim.x + threadIdx.x;   // over MTOT*24
  if (i < MTOT * 24) {
    const int t = i / 24;
    const int c = i - t * 24;
    yb[(size_t)t * DM + 72 + c] = 0.0f;
  }
}

// ---------------------------------------------------------------------------
// Dense a2a L1-attention over 150 selected tokens, head dim 6.
// One block per (batch, head); q/k/v tiles staged in LDS; one thread per query.
// Softmax denominator includes the phantom zero-key from reference padding.
// ---------------------------------------------------------------------------
__global__ void a2a_attn_kernel(const float* __restrict__ y,
                                const int* __restrict__ a2a,
                                float* __restrict__ yb)
{
  const int b = blockIdx.x >> 2;
  const int h = blockIdx.x & 3;
  __shared__ float qs[A2LEN * 6], ks[A2LEN * 6], vs[A2LEN * 6];

  for (int idx = threadIdx.x; idx < A2LEN * 6; idx += blockDim.x) {
    const int d = idx / 6;
    const int w = idx - d * 6;
    const int tok = a2a[d];
    const float* base = y + ((size_t)b * NTOK + tok) * (3 * DM);
    const int ch = 72 + h * 6 + w;
    qs[idx] = base[ch];
    ks[idx] = base[DM + ch];
    vs[idx] = base[2 * DM + ch];
  }
  __syncthreads();

  const int d = threadIdx.x;
  if (d >= A2LEN) return;

  const float scale = 0.40824829046386302f;  // 1/sqrt(6)
  float q0 = qs[d*6+0], q1 = qs[d*6+1], q2 = qs[d*6+2];
  float q3 = qs[d*6+3], q4 = qs[d*6+4], q5 = qs[d*6+5];

  // phantom zero-key score (index s == a2len in the padded reference)
  const float ph = -(fabsf(q0)+fabsf(q1)+fabsf(q2)+fabsf(q3)+fabsf(q4)+fabsf(q5)) * scale;
  float mx = ph;
  for (int s = 0; s < A2LEN; ++s) {
    const float* kp = ks + s * 6;
    const float sc = -(fabsf(q0-kp[0])+fabsf(q1-kp[1])+fabsf(q2-kp[2])
                      +fabsf(q3-kp[3])+fabsf(q4-kp[4])+fabsf(q5-kp[5])) * scale;
    mx = fmaxf(mx, sc);
  }
  float den = __expf(ph - mx);   // phantom contributes to denominator only
  float o0=0.f,o1=0.f,o2=0.f,o3=0.f,o4=0.f,o5=0.f;
  for (int s = 0; s < A2LEN; ++s) {
    const float* kp = ks + s * 6;
    const float sc = -(fabsf(q0-kp[0])+fabsf(q1-kp[1])+fabsf(q2-kp[2])
                      +fabsf(q3-kp[3])+fabsf(q4-kp[4])+fabsf(q5-kp[5])) * scale;
    const float e = __expf(sc - mx);
    den += e;
    const float* vp = vs + s * 6;
    o0 += e*vp[0]; o1 += e*vp[1]; o2 += e*vp[2];
    o3 += e*vp[3]; o4 += e*vp[4]; o5 += e*vp[5];
  }
  const float inv = 1.0f / den;
  const int tok = a2a[d];
  float* out = yb + ((size_t)b * NTOK + tok) * DM + 72 + h * 6;
  out[0]=o0*inv; out[1]=o1*inv; out[2]=o2*inv;
  out[3]=o3*inv; out[4]=o4*inv; out[5]=o5*inv;
}

// ---------------------------------------------------------------------------
// Sparse L1 attention, head dim 5, 32 neighbors per destination token.
// One block per (batch, token); thread = head*32 + slot, so each head's
// 32-slot softmax is exactly one wave32 -> shuffle reductions.
// ---------------------------------------------------------------------------
__global__ void sparse_attn_kernel(const float* __restrict__ y,
                                   const int* __restrict__ coo,  // (NTOK*32, 3): dst,src,cnt
                                   float* __restrict__ yb, int sta)
{
  const int bt = blockIdx.x;
  const int b  = bt >> 9;            // / NTOK
  const int t  = bt & (NTOK - 1);
  const int h  = threadIdx.x >> 5;
  const int r  = threadIdx.x & 31;

  const int e   = t * RMAX + r;      // the 32 entries with dst == t
  const int dst = coo[e * 3 + 0];    // == t by construction
  const int src = coo[e * 3 + 1];

  const float scale = 0.44721359549995793f;  // 1/sqrt(5)
  const float* qp = y + ((size_t)b * NTOK + dst) * (3 * DM) + sta + h * 5;
  const float* kp = y + ((size_t)b * NTOK + src) * (3 * DM) + DM + sta + h * 5;
  float s = 0.f;
  #pragma unroll
  for (int w = 0; w < 5; ++w) s += fabsf(qp[w] - kp[w]);
  const float ww = -s * scale;

  // wave-wide softmax over the 32 slots of this head
  float mx = ww;
  #pragma unroll
  for (int off = 16; off > 0; off >>= 1) mx = fmaxf(mx, __shfl_xor(mx, off, 32));
  const float ex = __expf(ww - mx);
  float den = ex;
  #pragma unroll
  for (int off = 16; off > 0; off >>= 1) den += __shfl_xor(den, off, 32);
  const float p = ex / den;

  const float* vp = y + ((size_t)b * NTOK + src) * (3 * DM) + 2 * DM + sta + h * 5;
  #pragma unroll
  for (int w = 0; w < 5; ++w) {
    float pv = p * vp[w];
    #pragma unroll
    for (int off = 16; off > 0; off >>= 1) pv += __shfl_xor(pv, off, 32);
    if (r == 0) yb[((size_t)b * NTOK + dst) * DM + sta + h * 5 + w] = pv;
  }
}

// ---------------------------------------------------------------------------
extern "C" void kernel_launch(void* const* d_in, const int* in_sizes, int n_in,
                              void* d_out, int out_size, void* d_ws, size_t ws_size,
                              hipStream_t stream) {
  const float* x    = (const float*)d_in[0];   // (32,512,96)
  const float* wqv  = (const float*)d_in[1];   // (288,97)
  const float* fan  = (const float*)d_in[2];   // (96,97)
  const int*   coo0 = (const int*)  d_in[3];   // (512*32,3)
  const int*   coo1 = (const int*)  d_in[4];
  const int*   a2a  = (const int*)  d_in[5];   // (150)
  float* out = (float*)d_out;                  // (32,512,96)

  float* y  = (float*)d_ws;                    // (16384, 288) qkv
  float* yb = y + (size_t)MTOT * 3 * DM;       // (16384, 96)  pre-gate concat

  // 1) QKV projection: y = x @ wqv[:, :96]^T + wqv[:, 96]
  //    Only q/k/v channels 32..95 are consumed downstream -> 12 of 18 N-tiles.
  {
    const int ntiles = 12;
    const int waves  = (MTOT / 16) * ntiles;   // 12288
    gemm_wmma_kernel<96, 0, DM, 97, 3 * DM, false, false, true>
        <<<waves / 8, 256, 0, stream>>>(x, wqv, nullptr, y, ntiles);
  }

  // 2) Zero dense-attention channels, then scatter a2a attention output.
  zero_dense_kernel<<<(MTOT * 24 + 255) / 256, 256, 0, stream>>>(yb);
  a2a_attn_kernel<<<BS * NH, 160, 0, stream>>>(y, a2a, yb);

  // 3) Two sparse L1 attentions (channels 32..51 and 52..71).
  sparse_attn_kernel<<<MTOT, 128, 0, stream>>>(y, coo0, yb, 32);
  sparse_attn_kernel<<<MTOT, 128, 0, stream>>>(y, coo1, yb, 52);

  // 4) Fanout: out = x + silu_gate(yb) @ fan[:, :96]^T + fan[:, 96]
  //    Channels 0..31 of gated yb are identically zero -> K loop starts at 32.
  {
    const int ntiles = DM / 16;                // 6
    const int waves  = (MTOT / 16) * ntiles;   // 6144
    gemm_wmma_kernel<96, 32, DM, 97, DM, true, true, false>
        <<<waves / 8, 256, 0, stream>>>(yb, fan, x, out, ntiles);
  }
}
